// MemTransformerLM_51771535786473
// MI455X (gfx1250) — compile-verified, matrix operand order checked by reading, not run on hardware
//
#include <hip/hip_runtime.h>

#define T_SEQ 2080
#define DM    512
#define HN    8
#define DH    64
#define DI    2048
#define NL    4

typedef _Float16 f16;
typedef __attribute__((ext_vector_type(16))) f16   v16h;
typedef __attribute__((ext_vector_type(8)))  float v8f;

union AFrag { v16h v; unsigned u[8]; };

__device__ __forceinline__ int kpair(int vv, int hf) {
  // K index of the low element of 32-bit pair vv for lane-half hf (16-bit A/B layout)
  return (vv < 4 ? 0 : 16) + hf * 8 + (vv & 3) * 2;
}

__device__ __forceinline__ v8f wmma_f16(const AFrag& a, const AFrag& b, v8f c) {
  return __builtin_amdgcn_wmma_f32_16x16x32_f16(false, a.v, false, b.v, (short)0, c,
                                                false, false);
}

__device__ __forceinline__ v8f vzero8() {
  v8f z = {0.f, 0.f, 0.f, 0.f, 0.f, 0.f, 0.f, 0.f};
  return z;
}

// Issue one wave's share of an (NI*8)x32-f16 tile copy (global -> LDS) as NI
// async 128-bit per-lane transfers (ASYNCcnt-tracked CDNA5 path).
// rowbase = wave_row_base + (lane>>2), cb = (lane&3)*16 bytes.
template <int NI>
__device__ __forceinline__ void async_tile(unsigned lds_base, const char* gbase,
                                           size_t row_stride_b, int rowbase, int cb) {
#pragma unroll
  for (int i = 0; i < NI; ++i) {
    int row = rowbase + i * 8;
    unsigned lds = lds_base + (unsigned)(row * 64 + cb);
    const char* g = gbase + (size_t)row * row_stride_b + cb;
    asm volatile("global_load_async_to_lds_b128 %0, %1, off" ::"v"(lds), "v"(g)
                 : "memory");
  }
}

// One K-step: wave tile 32(M) x 64(N): 2 A-frags x 4 B-frags -> 8 WMMAs.
__device__ __forceinline__ void gemm_step(const unsigned short* la,
                                          const unsigned short* lb, int wm, int wn,
                                          int l16, int hf, v8f (&acc)[2][4]) {
  AFrag af[2], bf[4];
#pragma unroll
  for (int mf = 0; mf < 2; ++mf) {
    int row = wm * 32 + mf * 16 + l16;
#pragma unroll
    for (int vv = 0; vv < 8; ++vv)
      af[mf].u[vv] = *(const unsigned*)&la[row * 32 + kpair(vv, hf)];
  }
#pragma unroll
  for (int nf = 0; nf < 4; ++nf) {
    int row = wn * 64 + nf * 16 + l16;
#pragma unroll
    for (int vv = 0; vv < 8; ++vv)
      bf[nf].u[vv] = *(const unsigned*)&lb[row * 32 + kpair(vv, hf)];
  }
#pragma unroll
  for (int mf = 0; mf < 2; ++mf)
#pragma unroll
    for (int nf = 0; nf < 4; ++nf)
      acc[mf][nf] = wmma_f16(af[mf], bf[nf], acc[mf][nf]);
}

// ---------------------------------------------------------------------------
// Generic WMMA GEMM:  C[M,N] = A16[M,K] * Bt16[N,K]^T  (+bias)(+relu)(+f16 copy)
// block = 128 threads (4 waves, 2x2), workgroup tile 64x128, wave tile 32x64,
// Kstep 32. Full-M tiles: double-buffered LDS fed by async global->LDS b128.
// ---------------------------------------------------------------------------
__global__ __launch_bounds__(128) void gemm_wmma_f16(
    const f16* __restrict__ A, const f16* __restrict__ Bt,
    float* __restrict__ C, f16* __restrict__ C16,
    const float* __restrict__ bias, int relu, int M, int N, int K) {
  __shared__ __attribute__((aligned(16))) unsigned short lsA[2][64 * 32];
  __shared__ __attribute__((aligned(16))) unsigned short lsB[2][128 * 32];

  const int tid  = threadIdx.x;
  const int lane = tid & 31, wv = tid >> 5;
  const int hf = lane >> 4, l16 = lane & 15;
  const int wm = wv >> 1, wn = wv & 1;
  const int n0 = blockIdx.x * 128, m0 = blockIdx.y * 64;
  const int K2 = K >> 1;
  const int nk = K >> 5;

  v8f acc[2][4];
  for (int i = 0; i < 2; ++i)
    for (int j = 0; j < 4; ++j) acc[i][j] = vzero8();

  if (m0 + 64 <= M) {
    // ---- fast path: async double-buffered tile pipeline ----
    const unsigned ldsA0 = (unsigned)(uintptr_t)&lsA[0][0];
    const unsigned ldsB0 = (unsigned)(uintptr_t)&lsB[0][0];
    const int rbA = wv * 16 + (lane >> 2);
    const int rbB = wv * 32 + (lane >> 2);
    const int cb = (lane & 3) * 16;
    const char* Ab = (const char*)A + (size_t)m0 * K * 2;
    const char* Bb = (const char*)Bt + (size_t)n0 * K * 2;

    async_tile<2>(ldsA0, Ab, (size_t)K * 2, rbA, cb);
    async_tile<4>(ldsB0, Bb, (size_t)K * 2, rbB, cb);
    for (int s = 0; s < nk; ++s) {
      int buf = s & 1;
      if (s + 1 < nk) {
        size_t koff = (size_t)(s + 1) * 64;  // 32 halfs = 64 bytes
        async_tile<2>(ldsA0 + (buf ^ 1) * 4096, Ab + koff, (size_t)K * 2, rbA, cb);
        async_tile<4>(ldsB0 + (buf ^ 1) * 8192, Bb + koff, (size_t)K * 2, rbB, cb);
        asm volatile("s_wait_asynccnt 0x6" ::: "memory");
      } else {
        asm volatile("s_wait_asynccnt 0x0" ::: "memory");
      }
      __syncthreads();
      gemm_step(lsA[buf], lsB[buf], wm, wn, l16, hf, acc);
      __syncthreads();
    }
  } else {
    // ---- edge path: guarded synchronous loads (single buffer) ----
    const unsigned* Au = (const unsigned*)A;
    const unsigned* Bu = (const unsigned*)Bt;
    for (int k0 = 0; k0 < K; k0 += 32) {
      __syncthreads();
      for (int idx = tid; idx < 1024; idx += 128) {
        int r = idx >> 4, c = idx & 15;
        unsigned av = 0;
        int rowA = m0 + r;
        if (rowA < M) av = Au[(size_t)rowA * K2 + (k0 >> 1) + c];
        ((unsigned*)lsA[0])[idx] = av;
      }
      for (int idx = tid; idx < 2048; idx += 128) {
        int r = idx >> 4, c = idx & 15;
        unsigned bv = 0;
        int rowB = n0 + r;
        if (rowB < N) bv = Bu[(size_t)rowB * K2 + (k0 >> 1) + c];
        ((unsigned*)lsB[0])[idx] = bv;
      }
      __syncthreads();
      gemm_step(lsA[0], lsB[0], wm, wn, l16, hf, acc);
    }
  }

  // epilogue
#pragma unroll
  for (int mf = 0; mf < 2; ++mf)
#pragma unroll
    for (int nf = 0; nf < 4; ++nf)
#pragma unroll
      for (int r = 0; r < 8; ++r) {
        int row = m0 + wm * 32 + mf * 16 + r + 8 * hf;
        int col = n0 + wn * 64 + nf * 16 + l16;
        if (row < M) {
          float v = acc[mf][nf][r];
          if (bias) v += bias[col];
          if (relu) v = fmaxf(v, 0.f);
          C[(size_t)row * N + col] = v;
          if (C16) C16[(size_t)row * N + col] = (f16)v;
        }
      }
}

// ---------------------------------------------------------------------------
// Fused Transformer-XL attention per (query block of 16, head).
// block = 256 threads (8 waves). Dynamic LDS:
//   [0, 133120)        float strip[16][2080]  (BD raw; later aliased by combine)
//   [133120, 141312)   f16 pstage[8][16][32]
// vt is V transposed: [H][DH][T] so PV B-fragments load as 32-bit pairs.
// ---------------------------------------------------------------------------
__global__ __launch_bounds__(256) void attn_wmma(
    const f16* __restrict__ qw, const f16* __restrict__ qr,
    const f16* __restrict__ kk, const f16* __restrict__ vt,
    const f16* __restrict__ rr, float* __restrict__ att,
    f16* __restrict__ att16) {
  extern __shared__ char smem[];
  float* strip = (float*)smem;
  unsigned short* ps = (unsigned short*)(smem + 16 * T_SEQ * 4);

  const int tid = threadIdx.x, lane = tid & 31, wav = tid >> 5;
  const int hf = lane >> 4, l16 = lane & 15;
  const int qb = blockIdx.x, h = blockIdx.y;
  const int i0 = qb * 16;
  const float scale = 0.125f;  // 1/sqrt(64)

  // ---- query fragments (shared by all key tiles) ----
  AFrag aqw[2], aqr[2];
  {
    size_t base = ((size_t)(i0 + l16) * HN + h) * DH;
#pragma unroll
    for (int s = 0; s < 2; ++s)
#pragma unroll
      for (int v = 0; v < 8; ++v) {
        int d = s * 32 + kpair(v, hf);
        aqw[s].u[v] = *(const unsigned*)(qw + base + d);
        aqr[s].u[v] = *(const unsigned*)(qr + base + d);
      }
  }

  // ---- Phase A: BD strip = (q + r_r_bias) . r^T  (16 x T) ----
  for (int jt = wav; jt < T_SEQ / 16; jt += 8) {
    AFrag bR[2];
    size_t jb = ((size_t)(jt * 16 + l16) * HN + h) * DH;
#pragma unroll
    for (int s = 0; s < 2; ++s)
#pragma unroll
      for (int v = 0; v < 8; ++v)
        bR[s].u[v] = *(const unsigned*)(rr + jb + s * 32 + kpair(v, hf));
    v8f c = vzero8();
    c = wmma_f16(aqr[0], bR[0], c);
    c = wmma_f16(aqr[1], bR[1], c);
#pragma unroll
    for (int r = 0; r < 8; ++r)
      strip[(r + 8 * hf) * T_SEQ + jt * 16 + l16] = c[r];
  }
  __syncthreads();

  // ---- Phase B: flash-attention over key chunks of 32 ----
  float m_run[8], s_run[8];
#pragma unroll
  for (int r = 0; r < 8; ++r) { m_run[r] = -1e30f; s_run[r] = 0.f; }
  v8f oacc[4];
#pragma unroll
  for (int nf = 0; nf < 4; ++nf) oacc[nf] = vzero8();

  for (int ct = wav; ct < T_SEQ / 32; ct += 8) {
    float sc[2][8];
#pragma unroll
    for (int st = 0; st < 2; ++st) {
      int j0 = ct * 32 + st * 16;
      AFrag bK[2];
      size_t kb = ((size_t)(j0 + l16) * HN + h) * DH;
#pragma unroll
      for (int s = 0; s < 2; ++s)
#pragma unroll
        for (int v = 0; v < 8; ++v)
          bK[s].u[v] = *(const unsigned*)(kk + kb + s * 32 + kpair(v, hf));
      v8f c = vzero8();
      c = wmma_f16(aqw[0], bK[0], c);
      c = wmma_f16(aqw[1], bK[1], c);
#pragma unroll
      for (int r = 0; r < 8; ++r) {
        int p = r + 8 * hf, i = i0 + p, j = j0 + l16;
        int cix = T_SEQ - 1 - i + j;
        cix = cix < 0 ? 0 : (cix > T_SEQ - 1 ? T_SEQ - 1 : cix);
        float bd = strip[p * T_SEQ + cix];
        float val = (c[r] + bd) * scale;
        bool keep = (j <= i) || (i < 16 && j < 16) ||
                    (i >= T_SEQ - 16 && j >= T_SEQ - 16);
        sc[st][r] = keep ? val : -1e30f;
      }
    }

    // online softmax update (row reductions across the 16 lanes of each half)
    float nm[8], alpha[8], p0[8], p1[8];
#pragma unroll
    for (int r = 0; r < 8; ++r) {
      float rm = fmaxf(sc[0][r], sc[1][r]);
      for (int o = 1; o < 16; o <<= 1) rm = fmaxf(rm, __shfl_xor(rm, o, 32));
      nm[r] = fmaxf(m_run[r], rm);
      alpha[r] = __expf(m_run[r] - nm[r]);
      m_run[r] = nm[r];
      p0[r] = (sc[0][r] <= -1e29f) ? 0.f : __expf(sc[0][r] - nm[r]);
      p1[r] = (sc[1][r] <= -1e29f) ? 0.f : __expf(sc[1][r] - nm[r]);
      float rs = p0[r] + p1[r];
      for (int o = 1; o < 16; o <<= 1) rs += __shfl_xor(rs, o, 32);
      s_run[r] = s_run[r] * alpha[r] + rs;
    }
#pragma unroll
    for (int nf = 0; nf < 4; ++nf)
#pragma unroll
      for (int r = 0; r < 8; ++r) oacc[nf][r] *= alpha[r];

    // stage P (C layout -> A layout) through wave-private LDS
    f16* pw = (f16*)(ps + wav * 512);
#pragma unroll
    for (int r = 0; r < 8; ++r) {
      pw[(r + 8 * hf) * 32 + l16]      = (f16)p0[r];
      pw[(r + 8 * hf) * 32 + 16 + l16] = (f16)p1[r];
    }
    asm volatile("" ::: "memory");  // same-wave LDS ops complete in order
    AFrag pf;
#pragma unroll
    for (int v = 0; v < 8; ++v)
      pf.u[v] = *(const unsigned*)&pw[l16 * 32 + kpair(v, hf)];

    // PV accumulate: oacc[nf] += P[16x32] * V[32x16(nf)] (Vt: pairs contiguous)
#pragma unroll
    for (int nf = 0; nf < 4; ++nf) {
      AFrag bV;
      const f16* vcol = vt + (size_t)(h * DH + nf * 16 + l16) * T_SEQ + ct * 32;
#pragma unroll
      for (int v = 0; v < 8; ++v)
        bV.u[v] = *(const unsigned*)(vcol + kpair(v, hf));
      oacc[nf] = wmma_f16(pf, bV, oacc[nf]);
    }
  }
  __syncthreads();  // strip no longer needed; alias combine buffers on top

  float* cm  = (float*)smem;        // [8][16]
  float* cs  = cm + 128;            // [8][16]
  float* cpv = cs + 128;            // [8][16][64]
  if (l16 == 0) {
#pragma unroll
    for (int r = 0; r < 8; ++r) {
      cm[wav * 16 + r + 8 * hf] = m_run[r];
      cs[wav * 16 + r + 8 * hf] = s_run[r];
    }
  }
#pragma unroll
  for (int nf = 0; nf < 4; ++nf)
#pragma unroll
    for (int r = 0; r < 8; ++r)
      cpv[(wav * 16 + r + 8 * hf) * 64 + nf * 16 + l16] = oacc[nf][r];
  __syncthreads();

  for (int idx = tid; idx < 16 * 64; idx += 256) {
    int p = idx >> 6, d = idx & 63;
    float M = -1e30f;
    for (int w = 0; w < 8; ++w) M = fmaxf(M, cm[w * 16 + p]);
    float S = 0.f, num = 0.f;
    for (int w = 0; w < 8; ++w) {
      float e = __expf(cm[w * 16 + p] - M);
      S += cs[w * 16 + p] * e;
      num += cpv[(w * 16 + p) * 64 + d] * e;
    }
    float o = num / S;
    size_t oi = (size_t)(i0 + p) * DM + h * DH + d;
    att[oi] = o;
    att16[oi] = (f16)o;
  }
}

// ---------------------------------------------------------------------------
// Elementwise kernels
// ---------------------------------------------------------------------------
__global__ void build_seq(const float* __restrict__ we, const float* __restrict__ mt,
                          float* __restrict__ w, f16* __restrict__ w16,
                          f16* __restrict__ pos16) {
  int idx = blockIdx.x * 256 + threadIdx.x;
  if (idx >= T_SEQ * DM) return;
  int t = idx >> 9, d = idx & 511;
  float val;
  if (t < 16) val = mt[t * DM + d];
  else if (t < T_SEQ - 16) val = we[(size_t)(t - 16) * DM + d];
  else val = mt[(t - (T_SEQ - 16)) * DM + d];
  w[idx] = val;
  w16[idx] = (f16)val;
  float pos = (float)(T_SEQ - 1 - t);
  int j = d & 255;
  float inv = __expf(-((float)(2 * j) * (1.f / 512.f)) * 9.2103403720f); // ln(1e4)
  float ang = pos * inv;
  pos16[idx] = (f16)((d < 256) ? __sinf(ang) : __cosf(ang));
}

__global__ void qkv_post(const float* __restrict__ heads,
                         const float* __restrict__ rwb, const float* __restrict__ rrb,
                         f16* __restrict__ qw, f16* __restrict__ qr,
                         f16* __restrict__ k16, f16* __restrict__ vt) {
  int idx = blockIdx.x * 256 + threadIdx.x;
  if (idx >= T_SEQ * DM) return;
  int t = idx >> 9, c = idx & 511;
  const float* row = heads + (size_t)t * (3 * DM);
  float q = row[c], k = row[DM + c], v = row[2 * DM + c];
  qw[idx] = (f16)(q + rwb[c]);
  qr[idx] = (f16)(q + rrb[c]);
  k16[idx] = (f16)k;
  vt[(size_t)c * T_SEQ + t] = (f16)v;  // transposed for PV fragments
}

__global__ __launch_bounds__(256) void ln_res(const float* __restrict__ x,
                                              const float* __restrict__ add,
                                              const float* __restrict__ g,
                                              const float* __restrict__ b,
                                              float* __restrict__ w,
                                              f16* __restrict__ w16) {
  __shared__ float sred[256];
  int t = blockIdx.x, tid = threadIdx.x;
  const float* xr = x + (size_t)t * DM;
  const float* ar = add + (size_t)t * DM;
  float v0 = xr[tid] + ar[tid];
  float v1 = xr[tid + 256] + ar[tid + 256];
  sred[tid] = v0 + v1;
  __syncthreads();
  for (int o = 128; o > 0; o >>= 1) {
    if (tid < o) sred[tid] += sred[tid + o];
    __syncthreads();
  }
  float mean = sred[0] * (1.f / 512.f);
  __syncthreads();
  float d0 = v0 - mean, d1 = v1 - mean;
  sred[tid] = d0 * d0 + d1 * d1;
  __syncthreads();
  for (int o = 128; o > 0; o >>= 1) {
    if (tid < o) sred[tid] += sred[tid + o];
    __syncthreads();
  }
  float inv = rsqrtf(sred[0] * (1.f / 512.f) + 1e-5f);
  float o0 = d0 * inv * g[tid] + b[tid];
  float o1 = d1 * inv * g[tid + 256] + b[tid + 256];
  size_t base = (size_t)t * DM;
  w[base + tid] = o0;
  w[base + tid + 256] = o1;
  w16[base + tid] = (f16)o0;
  w16[base + tid + 256] = (f16)o1;
}

__global__ void wtrans(const float* __restrict__ W, f16* __restrict__ Wt,
                       int K, int N) {
  int idx = blockIdx.x * 256 + threadIdx.x;
  if (idx >= K * N) return;
  int n = idx / K, k = idx - n * K;
  Wt[(size_t)n * K + k] = (f16)W[(size_t)k * N + n];
}

// ---------------------------------------------------------------------------
extern "C" void kernel_launch(void* const* d_in, const int* in_sizes, int n_in,
                              void* d_out, int out_size, void* d_ws, size_t ws_size,
                              hipStream_t stream) {
  (void)in_sizes; (void)n_in; (void)out_size; (void)ws_size;
  const float* word_emb = (const float*)d_in[0];
  const float* mem_tok  = (const float*)d_in[1];
  const float* Wqkv = (const float*)d_in[2];
  const float* Wr   = (const float*)d_in[3];
  const float* Wo   = (const float*)d_in[4];
  const float* ln1s = (const float*)d_in[5];
  const float* ln1b = (const float*)d_in[6];
  const float* W1   = (const float*)d_in[7];
  const float* b1   = (const float*)d_in[8];
  const float* W2   = (const float*)d_in[9];
  const float* b2   = (const float*)d_in[10];
  const float* ln2s = (const float*)d_in[11];
  const float* ln2b = (const float*)d_in[12];
  const float* rwb  = (const float*)d_in[13];
  const float* rrb  = (const float*)d_in[14];

  char* p = (char*)d_ws;
  auto alloc = [&](size_t bytes) {
    void* r = (void*)p;
    p += (bytes + 255) & ~(size_t)255;
    return r;
  };
  const size_t TD = (size_t)T_SEQ * DM;
  float* w     = (float*)alloc(TD * 4);
  f16*   w16   = (f16*)alloc(TD * 2);
  f16*   pos16 = (f16*)alloc(TD * 2);
  float* heads = (float*)alloc((size_t)T_SEQ * 3 * DM * 4);
  f16*   qw16  = (f16*)alloc(TD * 2);
  f16*   qr16  = (f16*)alloc(TD * 2);
  f16*   k16   = (f16*)alloc(TD * 2);
  f16*   vt16  = (f16*)alloc(TD * 2);
  f16*   r16   = (f16*)alloc(TD * 2);
  float* rtmp  = (float*)alloc(TD * 4);
  float* att   = (float*)alloc(TD * 4);
  f16*   att16 = (f16*)alloc(TD * 2);
  float* obuf  = (float*)alloc(TD * 4);
  float* ff1   = (float*)alloc((size_t)T_SEQ * DI * 4);
  f16*   ff116 = (f16*)alloc((size_t)T_SEQ * DI * 2);
  float* ff2   = (float*)alloc(TD * 4);
  f16* WqkvT = (f16*)alloc((size_t)NL * DM * 3 * DM * 2);
  f16* WrT   = (f16*)alloc((size_t)NL * DM * DM * 2);
  f16* WoT   = (f16*)alloc((size_t)NL * DM * DM * 2);
  f16* W1T   = (f16*)alloc((size_t)NL * DM * DI * 2);
  f16* W2T   = (f16*)alloc((size_t)NL * DI * DM * 2);

  // weight convert + transpose ([K,N] f32 -> [N,K] f16)
  for (int l = 0; l < NL; ++l) {
    wtrans<<<(DM * 3 * DM + 255) / 256, 256, 0, stream>>>(
        Wqkv + (size_t)l * DM * 3 * DM, WqkvT + (size_t)l * DM * 3 * DM, DM, 3 * DM);
    wtrans<<<(DM * DM + 255) / 256, 256, 0, stream>>>(
        Wr + (size_t)l * DM * DM, WrT + (size_t)l * DM * DM, DM, DM);
    wtrans<<<(DM * DM + 255) / 256, 256, 0, stream>>>(
        Wo + (size_t)l * DM * DM, WoT + (size_t)l * DM * DM, DM, DM);
    wtrans<<<(DM * DI + 255) / 256, 256, 0, stream>>>(
        W1 + (size_t)l * DM * DI, W1T + (size_t)l * DM * DI, DM, DI);
    wtrans<<<(DI * DM + 255) / 256, 256, 0, stream>>>(
        W2 + (size_t)l * DI * DM, W2T + (size_t)l * DI * DM, DI, DM);
  }

  build_seq<<<(T_SEQ * DM + 255) / 256, 256, 0, stream>>>(word_emb, mem_tok, w, w16,
                                                          pos16);

  const int MB = (T_SEQ + 63) / 64;  // 33
  const size_t ATTN_SMEM = (size_t)16 * T_SEQ * 4 + 8 * 16 * 32 * 2;  // 141312 B

  for (int l = 0; l < NL; ++l) {
    // heads = w @ Wqkv[l]
    gemm_wmma_f16<<<dim3(3 * DM / 128, MB), 128, 0, stream>>>(
        w16, WqkvT + (size_t)l * DM * 3 * DM, heads, nullptr, nullptr, 0,
        T_SEQ, 3 * DM, DM);
    // r = pos_emb @ Wr[l]  (f16 copy is what attention consumes)
    gemm_wmma_f16<<<dim3(DM / 128, MB), 128, 0, stream>>>(
        pos16, WrT + (size_t)l * DM * DM, rtmp, r16, nullptr, 0, T_SEQ, DM, DM);
    // split qkv, add rel-attn biases, cast (V stored transposed)
    qkv_post<<<(T_SEQ * DM + 255) / 256, 256, 0, stream>>>(heads, rwb, rrb, qw16,
                                                           qr16, k16, vt16);
    // fused rel-shift attention
    attn_wmma<<<dim3(T_SEQ / 16, HN), 256, ATTN_SMEM, stream>>>(qw16, qr16, k16,
                                                                vt16, r16, att,
                                                                att16);
    // o = attn_vec @ Wo[l]
    gemm_wmma_f16<<<dim3(DM / 128, MB), 128, 0, stream>>>(
        att16, WoT + (size_t)l * DM * DM, obuf, nullptr, nullptr, 0, T_SEQ, DM, DM);
    // w = LN(w + o)
    ln_res<<<T_SEQ, 256, 0, stream>>>(w, obuf, ln1s + l * DM, ln1b + l * DM, w, w16);
    // ff1 = relu(w @ W1 + b1)
    gemm_wmma_f16<<<dim3(DI / 128, MB), 128, 0, stream>>>(
        w16, W1T + (size_t)l * DM * DI, ff1, ff116, b1 + (size_t)l * DI, 1,
        T_SEQ, DI, DM);
    // ff2 = ff1 @ W2 + b2
    gemm_wmma_f16<<<dim3(DM / 128, MB), 128, 0, stream>>>(
        ff116, W2T + (size_t)l * DI * DM, ff2, nullptr, b2 + (size_t)l * DM, 0,
        T_SEQ, DM, DI);
    // w = LN(w + ff2): last layer writes straight to d_out
    float* wout = (l == NL - 1) ? (float*)d_out : w;
    ln_res<<<T_SEQ, 256, 0, stream>>>(w, ff2, ln2s + l * DM, ln2b + l * DM, wout,
                                      w16);
  }
}
